// Block_22651657519360
// MI455X (gfx1250) — compile-verified, matrix operand order checked by reading, not run on hardware
//
#include <hip/hip_runtime.h>

// ---------------------------------------------------------------------------
// CDNA5 (gfx1250, wave32) implementation of the equivariant GNN layer.
// - all channel-mixing matmuls via v_wmma_f32_16x16x32_bf16
// - GEMM->GEMM activations stored as bf16 (halves HBM traffic; HBM-bound)
// - bf16 A-tiles staged with global_load_async_to_lds_b128 (ASYNCcnt path)
// - dual-A fusion for  A1@W1 + A2@W2 (+bias+resid)  pairs
// ---------------------------------------------------------------------------

typedef __attribute__((ext_vector_type(16))) __bf16 v16bf;
typedef __attribute__((ext_vector_type(8)))  float  v8f;
typedef unsigned short ushortT;

#define NB    2
#define NN    10000
#define NE    160000
#define AD    128
#define VD    64
#define RK    32
#define EPS   1e-5f

#define FLAG_LEAKY 1
#define FLAG_ACC   2

static __device__ __forceinline__ ushortT f2bf(float f) {
  unsigned u = __float_as_uint(f);
  u += 0x7FFFu + ((u >> 16) & 1u);          // round-to-nearest-even
  return (ushortT)(u >> 16);
}
static __device__ __forceinline__ float bf2f(ushortT u) {
  return __uint_as_float(((unsigned)u) << 16);
}

// gather modes: 0 = direct; 1 = scalar gather (m=b*NE+e -> b*NN+idx[e]);
//               2 = vec gather (m=(b*NE+e)*3+c -> (b*NN+idx[e])*3+c)
template<int GM>
static __device__ __forceinline__ int arow_of(int m, const int* __restrict__ gidx) {
  if constexpr (GM == 0) {
    return m;
  } else if constexpr (GM == 1) {
    int b = m / NE, e = m - b * NE;
    return b * NN + gidx[e];
  } else {
    int er = m / 3, c3 = m - er * 3;
    int b = er / NE, e = er - b * NE;
    return (b * NN + gidx[e]) * 3 + c3;
  }
}

// Stage a 128xK A tile into LDS as bf16.
//  ABF=false: A is f32 -> load, convert, ds_store.
//  ABF=true : A is bf16 -> per-lane 16B DMA via global_load_async_to_lds_b128.
template<int K, int GM, bool ABF>
static __device__ __forceinline__ void stage_A(ushortT* sA, const void* __restrict__ Av,
                                               const int* __restrict__ gidx,
                                               int mBlock, int M) {
  if constexpr (!ABF) {
    const float* A = (const float*)Av;
    for (int i = threadIdx.x; i < 128 * K; i += 256) {
      int r = i / K, c = i - r * K;
      int m = mBlock + r;
      float v = 0.f;
      if (m < M) v = A[(size_t)arow_of<GM>(m, gidx) * K + c];
      sA[r * K + c] = f2bf(v);
    }
  } else {
    const ushortT* A = (const ushortT*)Av;
    constexpr int SEGS = K / 8;               // 16-byte segments per row
    for (int i = threadIdx.x; i < 128 * SEGS; i += 256) {
      int r = i / SEGS, sg = i - r * SEGS;
      int m = mBlock + r;
      ushortT* dst = sA + r * K + sg * 8;
      if (m < M) {
        const ushortT* src = A + (size_t)arow_of<GM>(m, gidx) * K + sg * 8;
        asm volatile("global_load_async_to_lds_b128 %0, %1, off"
                     :: "v"((unsigned)(size_t)dst), "v"(src) : "memory");
      } else {
        #pragma unroll
        for (int q = 0; q < 8; ++q) dst[q] = 0;
      }
    }
  }
}

template<int K, int N>
static __device__ __forceinline__ void stage_W(ushortT* sW,
                                               const float* __restrict__ W) {
  // store W transposed: sW[n][k] = W[k][n]  (contiguous K per column)
  for (int i = threadIdx.x; i < N * K; i += 256) {
    int n = i / K, k = i - n * K;
    sW[n * K + k] = f2bf(W[(size_t)k * N + n]);
  }
}

// ---------------------------------------------------------------------------
// Out[M,N] = op( A1[M,K]@W1[K,N]  [+ A2[M,K]@W2[K,N]]  [+bias] [+resid] )
// Block = 256 threads = 8 wave32; 128-row M tile; full K staged in LDS (bf16).
// ---------------------------------------------------------------------------
template<int K, int N, int GM1, int GM2, bool DUAL, bool ABF, bool OBF, bool RBF>
__launch_bounds__(256)
__global__ void gemm_tpl(const void* __restrict__ A1, const float* __restrict__ W1,
                         const void* __restrict__ A2, const float* __restrict__ W2,
                         const int* __restrict__ g1, const int* __restrict__ g2,
                         const float* __restrict__ bias, const void* __restrict__ resid,
                         void* __restrict__ Out, int M, int flags)
{
  constexpr int NA = DUAL ? 2 : 1;
  constexpr int NT = N / 16;
  __shared__ v16bf sAv[NA][128 * K / 16];   // 32B-aligned bf16 tiles
  __shared__ v16bf sWv[NA][N * K / 16];

  ushortT* sA0 = (ushortT*)&sAv[0][0];
  ushortT* sW0 = (ushortT*)&sWv[0][0];
  ushortT* sA1 = (ushortT*)&sAv[NA - 1][0];  // == sA0 when !DUAL
  ushortT* sW1 = (ushortT*)&sWv[NA - 1][0];

  const int tid    = threadIdx.x;
  const int mBlock = blockIdx.x * 128;

  if (tid < 32) __builtin_prefetch(W1 + tid * 8, 0, 0);    // global_prefetch_b8

  stage_A<K, GM1, ABF>(sA0, A1, g1, mBlock, M);
  stage_W<K, N>(sW0, W1);
  if constexpr (DUAL) {
    stage_A<K, GM2, ABF>(sA1, A2, g2, mBlock, M);
    stage_W<K, N>(sW1, W2);
  }
  if constexpr (ABF)
    asm volatile("s_wait_asynccnt 0x0" ::: "memory");      // own DMAs done
  __syncthreads();                                          // all waves done

  const int wave = tid >> 5;          // 8 waves of 32 (wave32)
  const int lane = tid & 31;
  const int hlf  = lane >> 4;
  const int l16  = lane & 15;

  v8f acc[NT] = {};
  const ushortT* a1p = sA0 + (wave * 16 + l16) * K + hlf * 16;
  const ushortT* a2p = sA1 + (wave * 16 + l16) * K + hlf * 16;
  const int bOff = l16 * K + hlf * 16;

  #pragma unroll
  for (int k0 = 0; k0 < K; k0 += 32) {
    v16bf af1 = *reinterpret_cast<const v16bf*>(a1p + k0);
    v16bf af2 = af1;                           // dead when !DUAL
    if constexpr (DUAL) af2 = *reinterpret_cast<const v16bf*>(a2p + k0);
    #pragma unroll
    for (int nt = 0; nt < NT; ++nt) {
      v16bf bf1 = *reinterpret_cast<const v16bf*>(sW0 + nt * 16 * K + bOff + k0);
      acc[nt] = __builtin_amdgcn_wmma_f32_16x16x32_bf16(
          false, af1, false, bf1, (short)0, acc[nt], false, false);
      if constexpr (DUAL) {
        v16bf bf2 = *reinterpret_cast<const v16bf*>(sW1 + nt * 16 * K + bOff + k0);
        acc[nt] = __builtin_amdgcn_wmma_f32_16x16x32_bf16(
            false, af2, false, bf2, (short)0, acc[nt], false, false);
      }
    }
  }

  // epilogue (doc layout: elem i of lane L -> M = i + 8*(L>>4), N = L&15)
  #pragma unroll
  for (int nt = 0; nt < NT; ++nt) {
    int col = nt * 16 + l16;
    float bv = bias ? bias[col] : 0.f;
    #pragma unroll
    for (int i = 0; i < 8; ++i) {
      int m = mBlock + wave * 16 + hlf * 8 + i;
      if (m >= M) continue;
      float v = acc[nt][i] + bv;
      size_t o = (size_t)m * N + col;
      if (resid) {
        if constexpr (RBF) v += bf2f(((const ushortT*)resid)[o]);
        else               v += ((const float*)resid)[o];
      }
      if (flags & FLAG_LEAKY) v = v > 0.f ? v : 0.2f * v;
      if (flags & FLAG_ACC) {
        if constexpr (OBF) v += bf2f(((const ushortT*)Out)[o]);
        else               v += ((const float*)Out)[o];
      }
      if constexpr (OBF) ((ushortT*)Out)[o] = f2bf(v);
      else               ((float*)Out)[o] = v;
    }
  }
}

template<int K, int N, int GM1, int GM2, bool DUAL, bool ABF, bool OBF, bool RBF>
static void launch_gemm(const void* A1, const float* W1,
                        const void* A2, const float* W2,
                        const int* g1, const int* g2,
                        const float* bias, const void* resid,
                        void* Out, long long M, int flags, hipStream_t s)
{
  gemm_tpl<K, N, GM1, GM2, DUAL, ABF, OBF, RBF>
      <<<dim3((unsigned)((M + 127) / 128)), dim3(256), 0, s>>>(
          A1, W1, A2, W2, g1, g2, bias, resid, Out, (int)M, flags);
}

// ---------------------------------------------------------------------------
// Elementwise / scatter kernels
// ---------------------------------------------------------------------------
__global__ void fillz(float* __restrict__ p, long long n)
{
  long long t = (long long)blockIdx.x * blockDim.x + threadIdx.x;
  if (t < n) p[t] = 0.f;
}

// symmetric neighbor scatter: Out[b,dst[e]] += X[b,src[e]]; Out[b,src[e]] += X[b,dst[e]]
__global__ void nbr_scatter(const float* __restrict__ X, float* __restrict__ Out,
                            const int* __restrict__ src, const int* __restrict__ dst,
                            int rowF)
{
  long long t = (long long)blockIdx.x * blockDim.x + threadIdx.x;
  long long total = (long long)NB * NE * rowF;
  if (t >= total) return;
  int c = (int)(t % rowF);
  long long t2 = t / rowF;
  int e = (int)(t2 % NE);
  int b = (int)(t2 / NE);
  int s = src[e], d = dst[e];
  float xs = X[((size_t)b * NN + s) * rowF + c];
  float xd = X[((size_t)b * NN + d) * rowF + c];
  atomicAdd(Out + ((size_t)b * NN + d) * rowF + c, xs);
  atomicAdd(Out + ((size_t)b * NN + s) * rowF + c, xd);
}

// edge->node scatter-add: Out[b,idx[e]] += T[b,e]
__global__ void scatter_add(const float* __restrict__ T, float* __restrict__ Out,
                            const int* __restrict__ idx, int rowF)
{
  long long t = (long long)blockIdx.x * blockDim.x + threadIdx.x;
  long long total = (long long)NB * NE * rowF;
  if (t >= total) return;
  int c = (int)(t % rowF);
  long long t2 = t / rowF;
  int e = (int)(t2 % NE);
  int b = (int)(t2 / NE);
  atomicAdd(Out + ((size_t)b * NN + idx[e]) * rowF + c,
            T[((size_t)b * NE + e) * rowF + c]);
}

// [row][64][3] f32 -> [row][3][64] f32
__global__ void vtrans_in(const float* __restrict__ Vin, float* __restrict__ Vout,
                          long long rows)
{
  long long t = (long long)blockIdx.x * blockDim.x + threadIdx.x;
  long long total = rows * 192;
  if (t >= total) return;
  long long row = t / 192;
  int i = (int)(t - row * 192);
  int ch = i / 3, c = i - ch * 3;
  Vout[row * 192 + c * 64 + ch] = Vin[t];
}

// [row][64][3] f32 -> [row][3][64] bf16
__global__ void vtrans_in_bf(const float* __restrict__ Vin, ushortT* __restrict__ Vout,
                             long long rows)
{
  long long t = (long long)blockIdx.x * blockDim.x + threadIdx.x;
  long long total = rows * 192;
  if (t >= total) return;
  long long row = t / 192;
  int i = (int)(t - row * 192);
  int ch = i / 3, c = i - ch * 3;
  Vout[row * 192 + c * 64 + ch] = f2bf(Vin[t]);
}

// v -> v * (1+|v|^2)^(-1/4), bf16 in-place, layout [row][3][64]
__global__ void vroot_bf(ushortT* __restrict__ V, long long rows)
{
  long long t = (long long)blockIdx.x * blockDim.x + threadIdx.x;
  long long total = rows * 64;
  if (t >= total) return;
  long long row = t / 64;
  int ch = (int)(t - row * 64);
  ushortT* p = V + row * 192 + ch;
  float a = bf2f(p[0]), b = bf2f(p[64]), c = bf2f(p[128]);
  float s = rsqrtf(sqrtf(1.f + a * a + b * b + c * c));
  p[0] = f2bf(a * s); p[64] = f2bf(b * s); p[128] = f2bf(c * s);
}

// dots[row][r] = sum_c L[row][c][r] * R[row][c][r]     (L,R: [row][3][32] f32)
__global__ void dots_k(const float* __restrict__ L, const float* __restrict__ R,
                       float* __restrict__ D, long long rows)
{
  long long t = (long long)blockIdx.x * blockDim.x + threadIdx.x;
  long long total = rows * 32;
  if (t >= total) return;
  long long row = t / 32;
  int r = (int)(t - row * 32);
  size_t b = (size_t)row * 96 + r;
  D[row * 32 + r] = L[b] * R[b] + L[b + 32] * R[b + 32] + L[b + 64] * R[b + 64];
}

// G[row][c][r] *= S[row][r]   (G: [row][3][32] f32)
__global__ void gate_k(float* __restrict__ G, const float* __restrict__ S,
                       long long rows)
{
  long long t = (long long)blockIdx.x * blockDim.x + threadIdx.x;
  long long total = rows * 96;
  if (t >= total) return;
  long long row = t / 96;
  int r = (int)(t % 32);
  G[t] *= S[row * 32 + r];
}

// per-channel sum & sumsq into stats[0..C) and stats[C..2C)
__launch_bounds__(256)
__global__ void bn_reduce(const float* __restrict__ X, float* __restrict__ stats,
                          long long rows, int C)
{
  __shared__ float ssum[256], ssq[256];
  int c = blockIdx.x;
  float s = 0.f, q = 0.f;
  for (long long r = (long long)blockIdx.y * 256 + threadIdx.x; r < rows;
       r += (long long)gridDim.y * 256) {
    float v = X[r * C + c];
    s += v; q += v * v;
  }
  ssum[threadIdx.x] = s; ssq[threadIdx.x] = q;
  __syncthreads();
  for (int k = 128; k > 0; k >>= 1) {
    if (threadIdx.x < k) {
      ssum[threadIdx.x] += ssum[threadIdx.x + k];
      ssq[threadIdx.x]  += ssq[threadIdx.x + k];
    }
    __syncthreads();
  }
  if (threadIdx.x == 0) {
    atomicAdd(&stats[c],     ssum[0]);
    atomicAdd(&stats[C + c], ssq[0]);
  }
}

// out = base + w*(z-mean)*rsqrt(var+eps) + b   (z may alias out: elementwise)
__global__ void apply_scal(const float* __restrict__ base, const float* __restrict__ Z,
                           const float* __restrict__ stats, const float* __restrict__ w,
                           const float* __restrict__ bb, float* __restrict__ out,
                           long long rows, int C)
{
  long long t = (long long)blockIdx.x * blockDim.x + threadIdx.x;
  long long total = rows * C;
  if (t >= total) return;
  int c = (int)(t % C);
  float mean = stats[c] / (float)rows;
  float var  = stats[C + c] / (float)rows - mean * mean;
  float z = Z[t];
  out[t] = base[t] + w[c] * (z - mean) * rsqrtf(var + EPS) + bb[c];
}

// out[row][ch][c] = base + scale[ch]*Z[row][c][ch]   (Z is a separate f32 buffer)
__global__ void apply_vec(const float* __restrict__ base, const float* __restrict__ Z,
                          const float* __restrict__ stats, const float* __restrict__ w,
                          float* __restrict__ out, long long rows)
{
  long long t = (long long)blockIdx.x * blockDim.x + threadIdx.x;
  long long total = rows * 192;
  if (t >= total) return;
  long long row = t / 192;
  int i = (int)(t - row * 192);
  int ch = i / 3, c = i - ch * 3;
  float n2 = stats[64 + ch] / ((float)rows * 3.f);
  float sc = w[ch] * rsqrtf(n2 + EPS);
  out[t] = base[t] + sc * Z[row * 192 + c * 64 + ch];
}

// in-place variant: out region currently holds Z as [row][3][64]; rewrite as
// base[row][64][3] + scale*Z.  Stage the row through LDS to transpose safely.
__launch_bounds__(192)
__global__ void apply_vec_inplace(const float* __restrict__ base,
                                  const float* __restrict__ stats,
                                  const float* __restrict__ w,
                                  float* __restrict__ out, long long rows)
{
  __shared__ float ld[192];
  long long row = blockIdx.x;
  int t = threadIdx.x;
  ld[t] = out[row * 192 + t];
  __syncthreads();
  int ch = t / 3, c = t - ch * 3;
  float n2 = stats[64 + ch] / ((float)rows * 3.f);
  float sc = w[ch] * rsqrtf(n2 + EPS);
  out[row * 192 + t] = base[row * 192 + t] + sc * ld[c * 64 + ch];
}

// ---------------------------------------------------------------------------
// Orchestration
// ---------------------------------------------------------------------------
extern "C" void kernel_launch(void* const* d_in, const int* in_sizes, int n_in,
                              void* d_out, int out_size, void* d_ws, size_t ws_size,
                              hipStream_t stream)
{
  (void)in_sizes; (void)n_in; (void)out_size; (void)ws_size;

  // inputs in setup_inputs() insertion order
  const float* a_node = (const float*)d_in[0];
  const float* v_node = (const float*)d_in[1];
  const float* a_edge = (const float*)d_in[2];
  const float* v_edge = (const float*)d_in[3];
  const int*   srcI   = (const int*)d_in[4];
  const int*   dstI   = (const int*)d_in[5];
  auto P = [&](int i) { return (const float*)d_in[i]; };
  const float *anc0_Ws = P(6),  *anc0_Wn = P(7),  *anc0_b = P(8);
  const float *vnc0_Ws = P(9),  *vnc0_Wn = P(10);
  const float *aer_Wsrc = P(11), *aer_Wdst = P(12), *aer_b = P(13);
  const float *ver_Wsrc = P(14), *ver_Wdst = P(15);
  const float *anl_W0 = P(16), *anl_b0 = P(17), *anl_W1 = P(18), *anl_b1 = P(19),
              *anl_W2 = P(20), *anl_b2 = P(21);
  const float *ael_W0 = P(22), *ael_b0 = P(23), *ael_W1 = P(24), *ael_b1 = P(25),
              *ael_W2 = P(26), *ael_b2 = P(27);
  const float *vnl_W0 = P(28), *vnl_W1 = P(29);
  const float *vel_W0 = P(30), *vel_W1 = P(31);
  const float *svp_Wl = P(32), *svp_Wr = P(33), *svp_Wa = P(34);
  const float *svp_Was = P(35), *svp_Wv = P(36), *svp_Wo = P(37);
  const float *aew_Wsrc = P(38), *aew_Wdst = P(39);
  const float *vew_Wsrc = P(40), *vew_Wdst = P(41);
  const float *anc1_Ws = P(42), *anc1_Wn = P(43), *anc1_b = P(44);
  const float *vnc1_Ws = P(45), *vnc1_Wn = P(46);
  const float *bn_an_w = P(47), *bn_an_b = P(48);
  const float *bn_ae_w = P(49), *bn_ae_b = P(50);
  const float *bn_vn_w = P(51), *bn_ve_w = P(52);

  const long long Rn = (long long)NB * NN;      // 20000 node rows
  const long long Re = (long long)NB * NE;      // 320000 edge rows

  // workspace layout (f32-sized slots; some slots hold bf16 data phases)
  float* w0 = (float*)d_ws;
  size_t off = 0;
  auto alloc = [&](long long n) { float* p = w0 + off; off += (size_t)n; return p; };
  float* nbrA  = alloc(Rn * AD);                 // f32 (atomics)
  float* xaF   = alloc(Rn * AD);                 // bf16 phase: xa
  float* tmpN1 = alloc(Rn * AD);                 // bf16 (mlp hidden) then f32 (conv1 out)
  float* tmpN2 = alloc(Rn * AD);                 // bf16 phase
  float* za    = alloc(Rn * AD);                 // f32 (atomics)
  float* vn3   = alloc(Rn * 192);                // f32
  float* nbrV3 = alloc(Rn * 192);                // f32 (atomics)
  float* xv3F  = alloc(Rn * 192);                // bf16 phase: xv3
  float* tmpNv = alloc(Rn * 192);                // bf16 (vmlp) then f32 (conv1 out)
  float* zv3   = alloc(Rn * 192);                // f32 (atomics)
  float* lN    = alloc(Rn * 96);                 // f32
  float* rN    = alloc(Rn * 96);                 // f32
  float* dotsN = alloc(Rn * 32);
  float* sN    = alloc(Rn * 32);
  float* statsAN = alloc(256);
  float* statsVN = alloc(128);
  float* statsAE = alloc(256);
  float* statsVE = alloc(128);
  float* xaeF  = alloc(Re * AD);                 // bf16 phase: xae
  float* tmpE1 = alloc(Re * AD);                 // bf16 (mlp hidden) then f32 (scatter tmp)
  float* tmpE2 = alloc(Re * AD);                 // bf16 phase
  float* xve3F = alloc(Re * 192);                // bf16 phase: xve3
  float* tmpEv = alloc(Re * 192);                // bf16 (vmlp) then f32 (scatter tmp)
  float* lE    = alloc(Re * 96);
  float* rE    = alloc(Re * 96);
  float* dotsE = alloc(Re * 32);
  float* sE    = alloc(Re * 32);

  ushortT* xa   = (ushortT*)xaF;
  ushortT* xv3  = (ushortT*)xv3F;
  ushortT* xae  = (ushortT*)xaeF;
  ushortT* xve3 = (ushortT*)xve3F;
  ushortT* h1N  = (ushortT*)tmpN1;
  ushortT* h2N  = (ushortT*)tmpN2;
  ushortT* h1E  = (ushortT*)tmpE1;
  ushortT* h2E  = (ushortT*)tmpE2;
  ushortT* hvN  = (ushortT*)tmpNv;
  ushortT* hvE  = (ushortT*)tmpEv;

  // d_out regions; z_ae / z_ve live directly in their output regions (f32)
  float* outAN = (float*)d_out;                 // [Rn,128]
  float* outVN = outAN + Rn * AD;               // [Rn,64,3]
  float* outAE = outVN + Rn * 192;              // [Re,128]  (zae scratch)
  float* outVE = outAE + Re * AD;               // [Re,192]  (zve3 scratch [er][3][64])

  auto ew = [&](long long total) { return dim3((unsigned)((total + 255) / 256)); };
  const int* NI = nullptr;
  const void* NV = nullptr;
  const float* NF = nullptr;

  // ---- 0) component-major transposes of vector inputs ----
  vtrans_in<<<ew(Rn * 192), 256, 0, stream>>>(v_node, vn3, Rn);
  vtrans_in_bf<<<ew(Re * 192), 256, 0, stream>>>(v_edge, xve3, Re); // resid (bf16)

  // ---- 1) neighbor sums for conv0 ----
  fillz<<<ew(Rn * AD), 256, 0, stream>>>(nbrA, Rn * AD);
  fillz<<<ew(Rn * 192), 256, 0, stream>>>(nbrV3, Rn * 192);
  nbr_scatter<<<ew((long long)NB * NE * AD), 256, 0, stream>>>(a_node, nbrA, srcI, dstI, AD);
  nbr_scatter<<<ew((long long)NB * NE * 192), 256, 0, stream>>>(vn3, nbrV3, srcI, dstI, 192);

  // ---- 2) conv0 (fused dual GEMMs; f32 in, bf16 out) ----
  launch_gemm<128,128,0,0,true ,false,true ,false>(a_node, anc0_Ws, nbrA, anc0_Wn,
      NI, NI, anc0_b, NV, xa, Rn, 0, stream);
  launch_gemm<64,64,0,0,true ,false,true ,false>(vn3, vnc0_Ws, nbrV3, vnc0_Wn,
      NI, NI, NF, NV, xv3, Rn * 3, 0, stream);

  // ---- 3) edges read (async bf16 gathers + residual + bias in one pass) ----
  launch_gemm<128,128,1,1,true ,true ,true ,false>(xa, aer_Wsrc, xa, aer_Wdst,
      srcI, dstI, aer_b, a_edge, xae, Re, 0, stream);
  launch_gemm<64,64,2,2,true ,true ,true ,true >(xv3, ver_Wsrc, xv3, ver_Wdst,
      srcI, dstI, NF, xve3, xve3, Re * 3, 0, stream);

  // ---- 4) node scalar MLP -> za ----
  launch_gemm<128,128,0,0,false,true ,true ,false>(xa,  anl_W0, NV, NF, NI, NI, anl_b0, NV, h1N, Rn, FLAG_LEAKY, stream);
  launch_gemm<128,128,0,0,false,true ,true ,false>(h1N, anl_W1, NV, NF, NI, NI, anl_b1, NV, h2N, Rn, FLAG_LEAKY, stream);
  launch_gemm<128,128,0,0,false,true ,false,false>(h2N, anl_W2, NV, NF, NI, NI, anl_b2, NV, za,  Rn, 0, stream);

  // ---- 5) node vector MLP -> zv3 ----
  launch_gemm<64,64,0,0,false,true ,true ,false>(xv3, vnl_W0, NV, NF, NI, NI, NF, NV, hvN, Rn * 3, 0, stream);
  vroot_bf<<<ew(Rn * VD), 256, 0, stream>>>(hvN, Rn);
  launch_gemm<64,64,0,0,false,true ,false,false>(hvN, vnl_W1, NV, NF, NI, NI, NF, NV, zv3, Rn * 3, 0, stream);

  // ---- 6) node scalar<->vector products ----
  launch_gemm<64,32,0,0,false,true ,false,false>(xv3, svp_Wl, NV, NF, NI, NI, NF, NV, lN, Rn * 3, 0, stream);
  launch_gemm<64,32,0,0,false,true ,false,false>(xv3, svp_Wr, NV, NF, NI, NI, NF, NV, rN, Rn * 3, 0, stream);
  dots_k<<<ew(Rn * RK), 256, 0, stream>>>(lN, rN, dotsN, Rn);
  launch_gemm<32,128,0,0,false,false,false,false>(dotsN, svp_Wa, NV, NF, NI, NI, NF, NV, za, Rn, FLAG_ACC, stream);
  launch_gemm<128,32,0,0,false,true ,false,false>(xa, svp_Was, NV, NF, NI, NI, NF, NV, sN, Rn, 0, stream);
  launch_gemm<64,32,0,0,false,true ,false,false>(xv3, svp_Wv, NV, NF, NI, NI, NF, NV, lN, Rn * 3, 0, stream);
  gate_k<<<ew(Rn * 96), 256, 0, stream>>>(lN, sN, Rn);
  launch_gemm<32,64,0,0,false,false,false,false>(lN, svp_Wo, NV, NF, NI, NI, NF, NV, zv3, Rn * 3, FLAG_ACC, stream);

  // ---- 7) edge scalar MLP -> zae (in outAE, f32) ----
  launch_gemm<128,128,0,0,false,true ,true ,false>(xae, ael_W0, NV, NF, NI, NI, ael_b0, NV, h1E, Re, FLAG_LEAKY, stream);
  launch_gemm<128,128,0,0,false,true ,true ,false>(h1E, ael_W1, NV, NF, NI, NI, ael_b1, NV, h2E, Re, FLAG_LEAKY, stream);
  launch_gemm<128,128,0,0,false,true ,false,false>(h2E, ael_W2, NV, NF, NI, NI, ael_b2, NV, outAE, Re, 0, stream);

  // ---- 8) edge vector MLP -> zve (in outVE, f32) ----
  launch_gemm<64,64,0,0,false,true ,true ,false>(xve3, vel_W0, NV, NF, NI, NI, NF, NV, hvE, Re * 3, 0, stream);
  vroot_bf<<<ew(Re * VD), 256, 0, stream>>>(hvE, Re);
  launch_gemm<64,64,0,0,false,true ,false,false>(hvE, vel_W1, NV, NF, NI, NI, NF, NV, outVE, Re * 3, 0, stream);

  // ---- 9) edge scalar<->vector products ----
  launch_gemm<64,32,0,0,false,true ,false,false>(xve3, svp_Wl, NV, NF, NI, NI, NF, NV, lE, Re * 3, 0, stream);
  launch_gemm<64,32,0,0,false,true ,false,false>(xve3, svp_Wr, NV, NF, NI, NI, NF, NV, rE, Re * 3, 0, stream);
  dots_k<<<ew(Re * RK), 256, 0, stream>>>(lE, rE, dotsE, Re);
  launch_gemm<32,128,0,0,false,false,false,false>(dotsE, svp_Wa, NV, NF, NI, NI, NF, NV, outAE, Re, FLAG_ACC, stream);
  launch_gemm<128,32,0,0,false,true ,false,false>(xae, svp_Was, NV, NF, NI, NI, NF, NV, sE, Re, 0, stream);
  launch_gemm<64,32,0,0,false,true ,false,false>(xve3, svp_Wv, NV, NF, NI, NI, NF, NV, lE, Re * 3, 0, stream);
  gate_k<<<ew(Re * 96), 256, 0, stream>>>(lE, sE, Re);
  launch_gemm<32,64,0,0,false,false,false,false>(lE, svp_Wo, NV, NF, NI, NI, NF, NV, outVE, Re * 3, FLAG_ACC, stream);

  // ---- 10) edges write: project (f32) then atomic scatter to nodes ----
  launch_gemm<128,128,0,0,false,false,false,false>(outAE, aew_Wsrc, NV, NF, NI, NI, NF, NV, tmpE1, Re, 0, stream);
  scatter_add<<<ew((long long)NB * NE * AD), 256, 0, stream>>>(tmpE1, za, srcI, AD);
  launch_gemm<128,128,0,0,false,false,false,false>(outAE, aew_Wdst, NV, NF, NI, NI, NF, NV, tmpE1, Re, 0, stream);
  scatter_add<<<ew((long long)NB * NE * AD), 256, 0, stream>>>(tmpE1, za, dstI, AD);
  launch_gemm<64,64,0,0,false,false,false,false>(outVE, vew_Wsrc, NV, NF, NI, NI, NF, NV, tmpEv, Re * 3, 0, stream);
  scatter_add<<<ew((long long)NB * NE * 192), 256, 0, stream>>>(tmpEv, zv3, srcI, 192);
  launch_gemm<64,64,0,0,false,false,false,false>(outVE, vew_Wdst, NV, NF, NI, NI, NF, NV, tmpEv, Re * 3, 0, stream);
  scatter_add<<<ew((long long)NB * NE * 192), 256, 0, stream>>>(tmpEv, zv3, dstI, 192);

  // ---- 11) conv1 (fused dual GEMMs, all f32) ----
  fillz<<<ew(Rn * AD), 256, 0, stream>>>(nbrA, Rn * AD);
  nbr_scatter<<<ew((long long)NB * NE * AD), 256, 0, stream>>>(za, nbrA, srcI, dstI, AD);
  launch_gemm<128,128,0,0,true ,false,false,false>(za, anc1_Ws, nbrA, anc1_Wn,
      NI, NI, anc1_b, NV, tmpN1, Rn, 0, stream);
  fillz<<<ew(Rn * 192), 256, 0, stream>>>(nbrV3, Rn * 192);
  nbr_scatter<<<ew((long long)NB * NE * 192), 256, 0, stream>>>(zv3, nbrV3, srcI, dstI, 192);
  launch_gemm<64,64,0,0,true ,false,false,false>(zv3, vnc1_Ws, nbrV3, vnc1_Wn,
      NI, NI, NF, NV, tmpNv, Rn * 3, 0, stream);

  // ---- 12) batch-norm statistics ----
  fillz<<<dim3(3), 256, 0, stream>>>(statsAN, 256 + 128 + 256 + 128);  // contiguous
  bn_reduce<<<dim3(AD, 64),  256, 0, stream>>>(tmpN1, statsAN, Rn, AD);
  bn_reduce<<<dim3(VD, 64),  256, 0, stream>>>(tmpNv, statsVN, Rn * 3, VD);
  bn_reduce<<<dim3(AD, 256), 256, 0, stream>>>(outAE, statsAE, Re, AD);
  bn_reduce<<<dim3(VD, 256), 256, 0, stream>>>(outVE, statsVE, Re * 3, VD);

  // ---- 13) BN + global residuals -> outputs ----
  apply_scal<<<ew(Rn * AD), 256, 0, stream>>>(a_node, tmpN1, statsAN, bn_an_w, bn_an_b,
                                              outAN, Rn, AD);
  apply_vec<<<ew(Rn * 192), 256, 0, stream>>>(v_node, tmpNv, statsVN, bn_vn_w, outVN, Rn);
  apply_scal<<<ew(Re * AD), 256, 0, stream>>>(a_edge, outAE, statsAE, bn_ae_w, bn_ae_b,
                                              outAE, Re, AD);
  apply_vec_inplace<<<dim3((unsigned)Re), 192, 0, stream>>>(v_edge, statsVE, bn_ve_w,
                                                            outVE, Re);
}
